// Decoder_8753143349713
// MI455X (gfx1250) — compile-verified, matrix operand order checked by reading, not run on hardware
//
#include <hip/hip_runtime.h>
#include <hip/hip_bf16.h>
#include <cstdint>
#include <cstddef>

// ---------------- problem constants ----------------
#define BB      32
#define T_ENC   256
#define EE      512
#define T_DEC   200
#define N_MELS  80
#define HH      1024
#define AA      128
#define KIN1    768     // 256 (prenet) + 512 (ctx)
#define KCAT    1536    // H + E

// ---------------- WMMA helpers (bf16 -> f32, 16x16x32) ----------------
typedef __attribute__((ext_vector_type(16))) __bf16 v16bf;
typedef __attribute__((ext_vector_type(8)))  float  v8f;

union Frag {
    unsigned short s[16];
    float4 q[2];
    v16bf v;
};

// Load one 16-element bf16 operand chunk for a lane:
// elements [0..7] at p, elements [8..15] at p+16 (K-striped per ISA layout).
__device__ __forceinline__ v16bf load_frag(const __hip_bfloat16* p) {
    Frag f;
    f.q[0] = *reinterpret_cast<const float4*>(p);
    f.q[1] = *reinterpret_cast<const float4*>(p + 16);
    return f.v;
}

// acc += A[16xK] * B[Kx16]; arow/brow already include row*ld + half*8.
__device__ __forceinline__ void wmma_loop(v8f& acc,
                                          const __hip_bfloat16* arow,
                                          const __hip_bfloat16* brow,
                                          int K) {
    for (int k = 0; k < K; k += 32) {
        v16bf a = load_frag(arow + k);
        v16bf b = load_frag(brow + k);
        acc = __builtin_amdgcn_wmma_f32_16x16x32_bf16(
            false, a, false, b, (short)0, acc, false, false);
    }
}

__device__ __forceinline__ float sigf(float x) {
    return 1.0f / (1.0f + __expf(-x));
}

__device__ __forceinline__ unsigned short f2bfbits(float x) {
    __hip_bfloat16 h = __float2bfloat16(x);
    unsigned short s;
    __builtin_memcpy(&s, &h, 2);
    return s;
}

// ---------------- utility kernels ----------------
__global__ void k_zero_u32(uint32_t* p, int n) {
    int i = blockIdx.x * 256 + threadIdx.x;
    if (i < n) p[i] = 0u;
}

__global__ void k_cvt_bf16(const float* src, __hip_bfloat16* dst, int n) {
    int i = blockIdx.x * 256 + threadIdx.x;
    if (i < n) dst[i] = __float2bfloat16(src[i]);
}

// src [K][N] row-major fp32 -> dst [N][Kpad] row-major bf16 (zero-padded K).
__global__ void k_transpose_bf16(const float* src, __hip_bfloat16* dst,
                                 int K, int N, int Kpad) {
    int i = blockIdx.x * 256 + threadIdx.x;
    if (i >= N * Kpad) return;
    int n = i / Kpad, k = i - n * Kpad;
    float v = (k < K) ? src[(size_t)k * N + n] : 0.0f;
    dst[i] = __float2bfloat16(v);
}

// ---------------- proc_enc = inputs @ W_enc  ([8192x512]x[512x128]) ----------------
__global__ __launch_bounds__(256) void k_proc_enc(const __hip_bfloat16* Abf,
                                                  const __hip_bfloat16* WencT,
                                                  float* C) {
    int tid = threadIdx.x, lane = tid & 31, wave = tid >> 5;
    int half = lane >> 4, lr = lane & 15;
    int id = blockIdx.x * 8 + wave;        // 4096 tiles
    int ntile = id & 7, mtile = id >> 3;   // 8 x 512
    const __hip_bfloat16* arow = Abf + (size_t)(mtile * 16 + lr) * EE + half * 8;
    const __hip_bfloat16* brow = WencT + (size_t)(ntile * 16 + lr) * EE + half * 8;
    v8f acc = {};
    wmma_loop(acc, arow, brow, EE);
    int r0 = mtile * 16 + half * 8, c0 = ntile * 16 + lr;
#pragma unroll
    for (int i = 0; i < 8; ++i)
        C[(size_t)(r0 + i) * AA + c0] = acc[i];
}

// ---------------- prenet: relu(relu(mel@pw1+b1)@pw2+b2) -> lstm_in[:, :256] ----------------
__global__ __launch_bounds__(256) void k_prenet(const float* padded_mels, int t,
                                                const __hip_bfloat16* pw1T, const float* pb1,
                                                const __hip_bfloat16* pw2T, const float* pb2,
                                                __hip_bfloat16* lstm_in) {
    __shared__ unsigned short melb[32 * 96];   // K padded 80 -> 96
    __shared__ unsigned short x1b[32 * 256];
    int tid = threadIdx.x;
    for (int i = tid; i < 32 * 96; i += 256) {
        int m = i / 96, k = i - m * 96;
        float v = 0.0f;
        if (t > 0 && k < N_MELS)
            v = padded_mels[(size_t)m * (N_MELS * T_DEC) + (size_t)k * T_DEC + (t - 1)];
        melb[i] = f2bfbits(v);
    }
    __syncthreads();
    int lane = tid & 31, wave = tid >> 5;
    int half = lane >> 4, lr = lane & 15;
    const __hip_bfloat16* mel = reinterpret_cast<const __hip_bfloat16*>(melb);
    // GEMM1: [32x96] x [96x256]
    for (int r = 0; r < 4; ++r) {
        int tile = wave + 8 * r;                 // 32 tiles
        int mtile = tile & 1, ntile = tile >> 1; // 2 x 16
        const __hip_bfloat16* arow = mel + (mtile * 16 + lr) * 96 + half * 8;
        const __hip_bfloat16* brow = pw1T + (size_t)(ntile * 16 + lr) * 96 + half * 8;
        v8f acc = {};
        wmma_loop(acc, arow, brow, 96);
        int r0 = mtile * 16 + half * 8, c0 = ntile * 16 + lr;
#pragma unroll
        for (int i = 0; i < 8; ++i) {
            float v = acc[i] + pb1[c0];
            v = v > 0.0f ? v : 0.0f;
            x1b[(r0 + i) * 256 + c0] = f2bfbits(v);
        }
    }
    __syncthreads();
    const __hip_bfloat16* x1 = reinterpret_cast<const __hip_bfloat16*>(x1b);
    // GEMM2: [32x256] x [256x256]
    for (int r = 0; r < 4; ++r) {
        int tile = wave + 8 * r;
        int mtile = tile & 1, ntile = tile >> 1;
        const __hip_bfloat16* arow = x1 + (mtile * 16 + lr) * 256 + half * 8;
        const __hip_bfloat16* brow = pw2T + (size_t)(ntile * 16 + lr) * 256 + half * 8;
        v8f acc = {};
        wmma_loop(acc, arow, brow, 256);
        int r0 = mtile * 16 + half * 8, c0 = ntile * 16 + lr;
#pragma unroll
        for (int i = 0; i < 8; ++i) {
            float v = acc[i] + pb2[c0];
            v = v > 0.0f ? v : 0.0f;
            lstm_in[(size_t)(r0 + i) * KIN1 + c0] = __float2bfloat16(v);
        }
    }
}

// ---------------- fused LSTM layer: gates GEMM (WMMA) + nonlinearity ----------------
// 64 blocks x 256 threads; block handles 16 output cols across all 4 gates.
__global__ __launch_bounds__(256) void k_lstm(const __hip_bfloat16* x, int Kx,
                                              const __hip_bfloat16* WiT,
                                              const __hip_bfloat16* h_prev,
                                              const __hip_bfloat16* WhT,
                                              const float* bias,
                                              float* c,
                                              __hip_bfloat16* h_out) {
    __shared__ float g[4][32][16];
    int tid = threadIdx.x, lane = tid & 31, wave = tid >> 5;
    int half = lane >> 4, lr = lane & 15;
    int nb = blockIdx.x * 16;
    int mtile = wave & 1, gate = wave >> 1;
    int col = gate * HH + nb;
    v8f acc = {};
    wmma_loop(acc, x + (size_t)(mtile * 16 + lr) * Kx + half * 8,
                   WiT + (size_t)(col + lr) * Kx + half * 8, Kx);
    wmma_loop(acc, h_prev + (size_t)(mtile * 16 + lr) * HH + half * 8,
                   WhT + (size_t)(col + lr) * HH + half * 8, HH);
    int r0 = mtile * 16 + half * 8;
#pragma unroll
    for (int i = 0; i < 8; ++i) g[gate][r0 + i][lr] = acc[i];
    __syncthreads();
    for (int idx = tid; idx < 512; idx += 256) {
        int m = idx >> 4, cc = idx & 15;
        int n = nb + cc;
        float gi = g[0][m][cc] + bias[n];
        float gf = g[1][m][cc] + bias[HH + n];
        float gg = g[2][m][cc] + bias[2 * HH + n];
        float go = g[3][m][cc] + bias[3 * HH + n];
        float cn = sigf(gf) * c[(size_t)m * HH + n] + sigf(gi) * tanhf(gg);
        c[(size_t)m * HH + n] = cn;
        h_out[(size_t)m * HH + n] = __float2bfloat16(sigf(go) * tanhf(cn));
    }
}

// ---------------- q = h2 @ W_q ([32x1024]x[1024x128]) ----------------
__global__ __launch_bounds__(256) void k_q(const __hip_bfloat16* h2,
                                           const __hip_bfloat16* WqT, float* q) {
    int tid = threadIdx.x, lane = tid & 31, wave = tid >> 5;
    int half = lane >> 4, lr = lane & 15;
    for (int r = 0; r < 2; ++r) {
        int tile = wave + 8 * r;                 // 16 tiles
        int mtile = tile & 1, ntile = tile >> 1; // 2 x 8
        v8f acc = {};
        wmma_loop(acc, h2 + (size_t)(mtile * 16 + lr) * HH + half * 8,
                       WqT + (size_t)(ntile * 16 + lr) * HH + half * 8, HH);
        int r0 = mtile * 16 + half * 8, c0 = ntile * 16 + lr;
#pragma unroll
        for (int i = 0; i < 8; ++i) q[(size_t)(r0 + i) * AA + c0] = acc[i];
    }
}

// ---------------- location-sensitive attention (1 block per batch row) ----------------
__global__ __launch_bounds__(256) void k_attn(const float* inputs, const float* proc_enc,
                                              const float* q, const float* conv_w,
                                              const float* conv_b, const float* W_loc,
                                              const float* v_e, float* aw_cum,
                                              __hip_bfloat16* lstm_in, __hip_bfloat16* ctxBf) {
    __shared__ float s_awc[T_ENC + 30];
    __shared__ float s_w[32 * 31];
    __shared__ float s_wloc[32 * AA];
    __shared__ float s_q[AA];
    __shared__ float s_ve[AA];
    __shared__ float s_cb[32];
    __shared__ float s_red[T_ENC];
    __shared__ float s_aw[T_ENC];
    int b = blockIdx.x, t = threadIdx.x;
    s_awc[15 + t] = aw_cum[b * T_ENC + t];
    if (t < 15) { s_awc[t] = 0.0f; s_awc[T_ENC + 15 + t] = 0.0f; }
    for (int i = t; i < 32 * 31; i += 256) s_w[i] = conv_w[i];
    for (int i = t; i < 32 * AA; i += 256) s_wloc[i] = W_loc[i];
    if (t < AA) { s_q[t] = q[b * AA + t]; s_ve[t] = v_e[t]; }
    if (t < 32) s_cb[t] = conv_b[t];
    __syncthreads();
    // 1D conv over cumulative attention (SAME, 31 taps, 32 channels)
    float co[32];
#pragma unroll
    for (int ch = 0; ch < 32; ++ch) {
        float a = s_cb[ch];
#pragma unroll
        for (int j = 0; j < 31; ++j) a += s_awc[t + j] * s_w[ch * 31 + j];
        co[ch] = a;
    }
    // energies
    const float* pe = proc_enc + ((size_t)b * T_ENC + t) * AA;
    float e = 0.0f;
    for (int a = 0; a < AA; ++a) {
        float l = 0.0f;
#pragma unroll
        for (int ch = 0; ch < 32; ++ch) l += co[ch] * s_wloc[ch * AA + a];
        e += tanhf(s_q[a] + l + pe[a]) * s_ve[a];
    }
    // softmax over T_ENC
    s_red[t] = e; __syncthreads();
    for (int off = 128; off > 0; off >>= 1) {
        if (t < off) s_red[t] = fmaxf(s_red[t], s_red[t + off]);
        __syncthreads();
    }
    float mx = s_red[0]; __syncthreads();
    float p = __expf(e - mx);
    s_red[t] = p; __syncthreads();
    for (int off = 128; off > 0; off >>= 1) {
        if (t < off) s_red[t] += s_red[t + off];
        __syncthreads();
    }
    float aw = p / s_red[0];
    s_aw[t] = aw;
    aw_cum[b * T_ENC + t] += aw;
    __syncthreads();
    // context = aw @ inputs[b]
    const float* inb = inputs + (size_t)b * T_ENC * EE;
    for (int rep = 0; rep < 2; ++rep) {
        int ed = t + rep * 256;
        float a = 0.0f;
        for (int tt = 0; tt < T_ENC; ++tt) a += s_aw[tt] * inb[(size_t)tt * EE + ed];
        __hip_bfloat16 hv = __float2bfloat16(a);
        lstm_in[(size_t)b * KIN1 + 256 + ed] = hv;  // feeds next step's LSTM1
        ctxBf[(size_t)b * EE + ed] = hv;            // feeds this step's projection
    }
}

// ---------------- output projection: [h2|ctx] @ Wp + bp -> mels[t] ----------------
__global__ __launch_bounds__(256) void k_out(const __hip_bfloat16* h2,
                                             const __hip_bfloat16* ctxBf,
                                             const __hip_bfloat16* WpT,
                                             const float* bp, float* out_t) {
    int tid = threadIdx.x, lane = tid & 31, wave = tid >> 5;
    int half = lane >> 4, lr = lane & 15;
    for (int tile = wave; tile < 10; tile += 8) {
        int mtile = tile & 1, ntile = tile >> 1;  // 2 x 5 (N=80)
        v8f acc = {};
        const __hip_bfloat16* brow = WpT + (size_t)(ntile * 16 + lr) * KCAT + half * 8;
        wmma_loop(acc, h2 + (size_t)(mtile * 16 + lr) * HH + half * 8, brow, HH);
        wmma_loop(acc, ctxBf + (size_t)(mtile * 16 + lr) * EE + half * 8, brow + HH, EE);
        int r0 = mtile * 16 + half * 8, c0 = ntile * 16 + lr;
#pragma unroll
        for (int i = 0; i < 8; ++i)
            out_t[(size_t)(r0 + i) * N_MELS + c0] = acc[i] + bp[c0];
    }
}

// ---------------- host launcher ----------------
extern "C" void kernel_launch(void* const* d_in, const int* in_sizes, int n_in,
                              void* d_out, int out_size, void* d_ws, size_t ws_size,
                              hipStream_t stream) {
    (void)in_sizes; (void)n_in; (void)out_size; (void)ws_size;
    const float* inputs      = (const float*)d_in[0];
    const float* padded_mels = (const float*)d_in[1];
    const float* W_enc       = (const float*)d_in[2];
    const float* W_q         = (const float*)d_in[3];
    const float* conv_w      = (const float*)d_in[4];
    const float* conv_b      = (const float*)d_in[5];
    const float* W_loc       = (const float*)d_in[6];
    const float* v_e         = (const float*)d_in[7];
    const float* pw1         = (const float*)d_in[8];
    const float* pb1         = (const float*)d_in[9];
    const float* pw2         = (const float*)d_in[10];
    const float* pb2         = (const float*)d_in[11];
    const float* Wi1         = (const float*)d_in[12];
    const float* Wh1         = (const float*)d_in[13];
    const float* bl1         = (const float*)d_in[14];
    const float* Wi2         = (const float*)d_in[15];
    const float* Wh2         = (const float*)d_in[16];
    const float* bl2         = (const float*)d_in[17];
    const float* Wp          = (const float*)d_in[18];
    const float* bp          = (const float*)d_in[19];
    float* mels = (float*)d_out;

    char* ws = (char*)d_ws;
    size_t o = 0;
    auto take = [&](size_t bytes) -> char* {
        char* p = ws + o;
        o = (o + bytes + 255) & ~(size_t)255;
        return p;
    };
    // weights (bf16, transposed to [N][Kpad])
    __hip_bfloat16* Wi1T  = (__hip_bfloat16*)take(4096ull * 768 * 2);
    __hip_bfloat16* Wh1T  = (__hip_bfloat16*)take(4096ull * 1024 * 2);
    __hip_bfloat16* Wi2T  = (__hip_bfloat16*)take(4096ull * 1024 * 2);
    __hip_bfloat16* Wh2T  = (__hip_bfloat16*)take(4096ull * 1024 * 2);
    __hip_bfloat16* pw1T  = (__hip_bfloat16*)take(256ull * 96 * 2);
    __hip_bfloat16* pw2T  = (__hip_bfloat16*)take(256ull * 256 * 2);
    __hip_bfloat16* WqT   = (__hip_bfloat16*)take(128ull * 1024 * 2);
    __hip_bfloat16* WencT = (__hip_bfloat16*)take(128ull * 512 * 2);
    __hip_bfloat16* WpT   = (__hip_bfloat16*)take(80ull * 1536 * 2);
    __hip_bfloat16* inBf  = (__hip_bfloat16*)take(32ull * 256 * 512 * 2);
    float* procEnc        = (float*)take(32ull * 256 * 128 * 4);
    // state (zeroed every call) -- keep contiguous for a single zero kernel
    char* stateBegin = ws + o;
    float* c1    = (float*)take(32ull * 1024 * 4);
    float* c2    = (float*)take(32ull * 1024 * 4);
    __hip_bfloat16* h1p0 = (__hip_bfloat16*)take(32ull * 1024 * 2);
    __hip_bfloat16* h1p1 = (__hip_bfloat16*)take(32ull * 1024 * 2);
    __hip_bfloat16* h2p0 = (__hip_bfloat16*)take(32ull * 1024 * 2);
    __hip_bfloat16* h2p1 = (__hip_bfloat16*)take(32ull * 1024 * 2);
    __hip_bfloat16* lstmIn = (__hip_bfloat16*)take(32ull * KIN1 * 2);
    __hip_bfloat16* ctxBf  = (__hip_bfloat16*)take(32ull * 512 * 2);
    float* awCum = (float*)take(32ull * 256 * 4);
    float* qbuf  = (float*)take(32ull * 128 * 4);
    size_t stateBytes = (size_t)((ws + o) - stateBegin);

    auto g = [](size_t n) { return (unsigned)((n + 255) / 256); };

    // one-time (per launch) weight prep
    k_transpose_bf16<<<g(4096ull * 768), 256, 0, stream>>>(Wi1, Wi1T, 768, 4096, 768);
    k_transpose_bf16<<<g(4096ull * 1024), 256, 0, stream>>>(Wh1, Wh1T, 1024, 4096, 1024);
    k_transpose_bf16<<<g(4096ull * 1024), 256, 0, stream>>>(Wi2, Wi2T, 1024, 4096, 1024);
    k_transpose_bf16<<<g(4096ull * 1024), 256, 0, stream>>>(Wh2, Wh2T, 1024, 4096, 1024);
    k_transpose_bf16<<<g(256ull * 96), 256, 0, stream>>>(pw1, pw1T, 80, 256, 96);
    k_transpose_bf16<<<g(256ull * 256), 256, 0, stream>>>(pw2, pw2T, 256, 256, 256);
    k_transpose_bf16<<<g(128ull * 1024), 256, 0, stream>>>(W_q, WqT, 1024, 128, 1024);
    k_transpose_bf16<<<g(128ull * 512), 256, 0, stream>>>(W_enc, WencT, 512, 128, 512);
    k_transpose_bf16<<<g(80ull * 1536), 256, 0, stream>>>(Wp, WpT, 1536, 80, 1536);
    k_cvt_bf16<<<g(32ull * 256 * 512), 256, 0, stream>>>(inputs, inBf, 32 * 256 * 512);
    k_zero_u32<<<g(stateBytes / 4), 256, 0, stream>>>((uint32_t*)stateBegin, (int)(stateBytes / 4));
    k_proc_enc<<<512, 256, 0, stream>>>(inBf, WencT, procEnc);

    // sequential decode
    for (int t = 0; t < T_DEC; ++t) {
        __hip_bfloat16* h1prev = (t & 1) ? h1p1 : h1p0;
        __hip_bfloat16* h1cur  = (t & 1) ? h1p0 : h1p1;
        __hip_bfloat16* h2prev = (t & 1) ? h2p1 : h2p0;
        __hip_bfloat16* h2cur  = (t & 1) ? h2p0 : h2p1;
        k_prenet<<<1, 256, 0, stream>>>(padded_mels, t, pw1T, pb1, pw2T, pb2, lstmIn);
        k_lstm<<<64, 256, 0, stream>>>(lstmIn, KIN1, Wi1T, h1prev, Wh1T, bl1, c1, h1cur);
        k_lstm<<<64, 256, 0, stream>>>(h1cur, HH, Wi2T, h2prev, Wh2T, bl2, c2, h2cur);
        k_q<<<1, 256, 0, stream>>>(h2cur, WqT, qbuf);
        k_attn<<<32, 256, 0, stream>>>(inputs, procEnc, qbuf, conv_w, conv_b,
                                       W_loc, v_e, awCum, lstmIn, ctxBf);
        k_out<<<1, 256, 0, stream>>>(h2cur, ctxBf, WpT, bp,
                                     mels + (size_t)t * BB * N_MELS);
    }
}
